// Generator_71038759076226
// MI455X (gfx1250) — matrix operand order repaired and checked by reading, not compiled, not run on hardware
//
#include <hip/hip_runtime.h>
#include <hip/hip_bf16.h>
#include <math.h>

// ---------------------------------------------------------------------------
// LSTM greedy decoder, 20 steps. Bandwidth-bound GEMVs -> bf16 weights + WMMA.
// ---------------------------------------------------------------------------

typedef __attribute__((ext_vector_type(16))) __bf16 v16bf;
typedef __attribute__((ext_vector_type(8)))  float  v8f;

#define HDIM   4096
#define GDIM   16384          // 4*H
#define VDIM   32000

union Frag {                  // 32 bytes: a 16-element bf16 WMMA fragment
    v16bf v;
    uint4 q[2];
};

__device__ __forceinline__ unsigned short f2bf(float x) {
    // round-to-nearest-even fp32 -> bf16
    unsigned u = __float_as_uint(x);
    unsigned r = u + 0x7FFFu + ((u >> 16) & 1u);
    return (unsigned short)(r >> 16);
}

__device__ __forceinline__ float sigm(float x) { return 1.0f / (1.0f + expf(-x)); }

// monotone mapping float -> uint for atomicMax-based argmax
__device__ __forceinline__ unsigned fkey(float x) {
    unsigned u = __float_as_uint(x);
    return (u & 0x80000000u) ? ~u : (u | 0x80000000u);
}

// ---------------------------------------------------------------------------
// fp32 -> bf16 weight conversion (vectorized, grid-stride)
// ---------------------------------------------------------------------------
__global__ __launch_bounds__(256) void cvt_bf16_kernel(const float* __restrict__ src,
                                                       unsigned short* __restrict__ dst,
                                                       long long n4) {
    const float4* s4 = (const float4*)src;
    ushort4*      d4 = (ushort4*)dst;
    long long i = (long long)blockIdx.x * blockDim.x + threadIdx.x;
    long long stride = (long long)gridDim.x * blockDim.x;
    for (; i < n4; i += stride) {
        float4 v = s4[i];
        ushort4 o;
        o.x = f2bf(v.x); o.y = f2bf(v.y); o.z = f2bf(v.z); o.w = f2bf(v.w);
        d4[i] = o;
    }
}

// ---------------------------------------------------------------------------
// init: h0 = noise + image_features (as bf16), c0 = 0, argmax slot -> token 0
// ---------------------------------------------------------------------------
__global__ __launch_bounds__(256) void init_kernel(const float* __restrict__ img,
                                                   const float* __restrict__ noise,
                                                   unsigned short* __restrict__ hbf,
                                                   float* __restrict__ c,
                                                   unsigned long long* __restrict__ amax) {
    int i = blockIdx.x * blockDim.x + threadIdx.x;
    if (i < HDIM) {
        hbf[i] = f2bf(img[i] + noise[i]);
        c[i]   = 0.0f;
    }
    if (i == 0) *amax = ~0ull;   // low 32 bits 0xFFFFFFFF -> decodes to token 0
}

// ---------------------------------------------------------------------------
// gates = W_ih @ x + W_hh @ h + b_ih + b_hh
// One wave per 16-row tile. A = weight tile (per ISA bf16 A-layout),
// B = vector broadcast over all 16 columns (read from LDS).
// ---------------------------------------------------------------------------
__global__ __launch_bounds__(128) void gates_kernel(const float* __restrict__ emb,
                                                    const unsigned short* __restrict__ wih,
                                                    const unsigned short* __restrict__ whh,
                                                    const float* __restrict__ bih,
                                                    const float* __restrict__ bhh,
                                                    const unsigned short* __restrict__ hbf,
                                                    const unsigned long long* __restrict__ amax,
                                                    float* __restrict__ gates) {
    __shared__ __align__(16) unsigned short lx[HDIM];
    __shared__ __align__(16) unsigned short lh[HDIM];

    const int tid = threadIdx.x;

    // decode greedy token from packed argmax slot
    unsigned tok = 0xFFFFFFFFu - (unsigned)(*amax & 0xFFFFFFFFull);
    if (tok >= (unsigned)VDIM) tok = 0;
    const float* erow = emb + (size_t)tok * HDIM;

    for (int i = tid; i < HDIM; i += 128) {
        lx[i] = f2bf(erow[i]);   // embedding row fp32 -> bf16
        lh[i] = hbf[i];
    }
    __syncthreads();

    const int wave = tid >> 5;
    const int lane = tid & 31;
    const int r    = lane & 15;
    const int hi   = lane >> 4;                       // 0: lanes 0-15, 1: lanes 16-31
    const int row0 = (blockIdx.x * 4 + wave) * 16;

    v8f acc = {};

    // ---- W_ih @ x ----
    {
        const char* pa = (const char*)(wih + (size_t)(row0 + r) * HDIM) + hi * 16;
        const char* pb = (const char*)lx + hi * 16;
#pragma unroll 4
        for (int k = 0; k < HDIM; k += 32) {
            Frag a, b;
            a.q[0] = *(const uint4*)(pa);
            a.q[1] = *(const uint4*)(pa + 32);
            b.q[0] = *(const uint4*)(pb);
            b.q[1] = *(const uint4*)(pb + 32);
            acc = __builtin_amdgcn_wmma_f32_16x16x32_bf16(false, a.v, false, b.v,
                                                          (short)0, acc, false, false);
            pa += 64; pb += 64;
        }
    }
    // ---- + W_hh @ h ----
    {
        const char* pa = (const char*)(whh + (size_t)(row0 + r) * HDIM) + hi * 16;
        const char* pb = (const char*)lh + hi * 16;
#pragma unroll 4
        for (int k = 0; k < HDIM; k += 32) {
            Frag a, b;
            a.q[0] = *(const uint4*)(pa);
            a.q[1] = *(const uint4*)(pa + 32);
            b.q[0] = *(const uint4*)(pb);
            b.q[1] = *(const uint4*)(pb + 32);
            acc = __builtin_amdgcn_wmma_f32_16x16x32_bf16(false, a.v, false, b.v,
                                                          (short)0, acc, false, false);
            pa += 64; pb += 64;
        }
    }

    // D: all columns identical; column 0 lives in lane 0 (M=0..7) and lane 16 (M=8..15)
    if (lane == 0 || lane == 16) {
        int base = row0 + hi * 8;
#pragma unroll
        for (int m = 0; m < 8; ++m)
            gates[base + m] = acc[m] + bih[base + m] + bhh[base + m];
    }
}

// ---------------------------------------------------------------------------
// LSTM cell update (torch gate order i,f,g,o); also resets argmax slot
// ---------------------------------------------------------------------------
__global__ __launch_bounds__(256) void cell_kernel(const float* __restrict__ gates,
                                                   float* __restrict__ c,
                                                   unsigned short* __restrict__ hbf,
                                                   unsigned long long* __restrict__ amax) {
    int i = blockIdx.x * blockDim.x + threadIdx.x;
    if (i == 0) *amax = 0ull;
    if (i < HDIM) {
        float gi = gates[i];
        float gf = gates[HDIM + i];
        float gg = gates[2 * HDIM + i];
        float go = gates[3 * HDIM + i];
        float cc = sigm(gf) * c[i] + sigm(gi) * tanhf(gg);
        c[i] = cc;
        hbf[i] = f2bf(sigm(go) * tanhf(cc));
    }
}

// ---------------------------------------------------------------------------
// logits = W_out @ h + b_out ; fused greedy argmax via block reduce + atomicMax
// ---------------------------------------------------------------------------
__global__ __launch_bounds__(128) void logits_kernel(const unsigned short* __restrict__ wout,
                                                     const unsigned short* __restrict__ hbf,
                                                     const float* __restrict__ bout,
                                                     float* __restrict__ out,
                                                     unsigned long long* __restrict__ amax) {
    __shared__ __align__(16) unsigned short lh[HDIM];
    __shared__ unsigned long long red[8];

    const int tid = threadIdx.x;
    for (int i = tid; i < HDIM; i += 128) lh[i] = hbf[i];
    __syncthreads();

    const int wave = tid >> 5;
    const int lane = tid & 31;
    const int r    = lane & 15;
    const int hi   = lane >> 4;
    const int row0 = (blockIdx.x * 4 + wave) * 16;   // 500 blocks * 4 * 16 = 32000

    v8f acc = {};
    {
        const char* pa = (const char*)(wout + (size_t)(row0 + r) * HDIM) + hi * 16;
        const char* pb = (const char*)lh + hi * 16;
#pragma unroll 4
        for (int k = 0; k < HDIM; k += 32) {
            Frag a, b;
            a.q[0] = *(const uint4*)(pa);
            a.q[1] = *(const uint4*)(pa + 32);
            b.q[0] = *(const uint4*)(pb);
            b.q[1] = *(const uint4*)(pb + 32);
            acc = __builtin_amdgcn_wmma_f32_16x16x32_bf16(false, a.v, false, b.v,
                                                          (short)0, acc, false, false);
            pa += 64; pb += 64;
        }
    }

    if (lane == 0 || lane == 16) {
        int base = row0 + hi * 8;
        float best = -__builtin_inff();
        int   bidx = base;
#pragma unroll
        for (int m = 0; m < 8; ++m) {
            float v = acc[m] + bout[base + m];
            out[base + m] = v;
            if (v > best) { best = v; bidx = base + m; }   // strict > keeps first max
        }
        // pack: high 32 = monotone float key, low 32 = ~index (smaller index wins ties)
        red[wave * 2 + hi] = ((unsigned long long)fkey(best) << 32) |
                             (unsigned long long)(0xFFFFFFFFu - (unsigned)bidx);
    }
    __syncthreads();
    if (tid == 0) {
        unsigned long long m = red[0];
#pragma unroll
        for (int i = 1; i < 8; ++i) m = (red[i] > m) ? red[i] : m;
        atomicMax(amax, m);
    }
}

// ---------------------------------------------------------------------------
// host-side launcher
// ---------------------------------------------------------------------------
extern "C" void kernel_launch(void* const* d_in, const int* in_sizes, int n_in,
                              void* d_out, int out_size, void* d_ws, size_t ws_size,
                              hipStream_t stream) {
    (void)in_sizes; (void)n_in; (void)out_size; (void)ws_size;

    const float* img   = (const float*)d_in[0];   // [1,1,4096]
    const float* noise = (const float*)d_in[1];   // [1,1,4096]
    const float* emb   = (const float*)d_in[2];   // [32000,4096]
    const float* Wih   = (const float*)d_in[3];   // [16384,4096]
    const float* Whh   = (const float*)d_in[4];   // [16384,4096]
    const float* bih   = (const float*)d_in[5];   // [16384]
    const float* bhh   = (const float*)d_in[6];   // [16384]
    const float* Wout  = (const float*)d_in[7];   // [32000,4096]
    const float* bout  = (const float*)d_in[8];   // [32000]
    float*       out   = (float*)d_out;           // [20,32000]

    // workspace carve (needs ~507 MB)
    const size_t N_IH  = (size_t)GDIM * HDIM;     // 67,108,864
    const size_t N_OUT = (size_t)VDIM * HDIM;     // 131,072,000
    char* w = (char*)d_ws;
    unsigned short* wih_bf  = (unsigned short*)w;  w += N_IH * 2;
    unsigned short* whh_bf  = (unsigned short*)w;  w += N_IH * 2;
    unsigned short* wout_bf = (unsigned short*)w;  w += N_OUT * 2;
    unsigned short* h_bf    = (unsigned short*)w;  w += (size_t)HDIM * 2;
    float*          c_st    = (float*)w;           w += (size_t)HDIM * 4;
    float*          gates   = (float*)w;           w += (size_t)GDIM * 4;
    unsigned long long* amax = (unsigned long long*)w;

    // one-time (per call) fp32 -> bf16 weight conversion: halves streamed bytes
    cvt_bf16_kernel<<<4096, 256, 0, stream>>>(Wih,  wih_bf,  (long long)(N_IH  / 4));
    cvt_bf16_kernel<<<4096, 256, 0, stream>>>(Whh,  whh_bf,  (long long)(N_IH  / 4));
    cvt_bf16_kernel<<<8192, 256, 0, stream>>>(Wout, wout_bf, (long long)(N_OUT / 4));

    init_kernel<<<(HDIM + 255) / 256, 256, 0, stream>>>(img, noise, h_bf, c_st, amax);

    for (int t = 0; t < 20; ++t) {
        gates_kernel<<<GDIM / 64, 128, 0, stream>>>(emb, wih_bf, whh_bf, bih, bhh,
                                                    h_bf, amax, gates);
        cell_kernel<<<(HDIM + 255) / 256, 256, 0, stream>>>(gates, c_st, h_bf, amax);
        logits_kernel<<<VDIM / 64, 128, 0, stream>>>(wout_bf, h_bf, bout,
                                                     out + (size_t)t * VDIM, amax);
    }
}
